// TransformerCrossBlock_66468913873004
// MI455X (gfx1250) — matrix operand order repaired and checked by reading, not compiled, not run on hardware
//
#include <hip/hip_runtime.h>

// Problem constants
#define B_  16
#define N_  2048
#define C_  1024
#define H_  16
#define HD_ 64
#define BN_ (B_ * N_)

typedef unsigned short u16;
typedef __attribute__((ext_vector_type(16))) __bf16 v16bf;
typedef __attribute__((ext_vector_type(8)))  float  v8f;

union BF16Frag { v16bf v; unsigned u[8]; };
static_assert(sizeof(v16bf) == 32, "v16bf must be 8 VGPRs");

__device__ __forceinline__ unsigned f2bf(float f) {
  unsigned u = __float_as_uint(f);
  return (u + 0x7FFFu + ((u >> 16) & 1u)) >> 16;   // round-to-nearest-even
}
__device__ __forceinline__ float bf2f(unsigned h) { return __uint_as_float(h << 16); }

// ---------------------------------------------------------------------------
// CDNA5 async global->LDS DMA (ASYNCcnt-tracked, no VGPR round trip).
// LDS byte address = low 32 bits of the generic pointer (ISA 10.2: aperture
// occupies addr[63:32], addr[31:0] is the LDS offset).
// ---------------------------------------------------------------------------
__device__ __forceinline__ void async_b128(const u16* gsrc, u16* ldst) {
  asm volatile("global_load_async_to_lds_b128 %0, %1, off"
               :
               : "v"((unsigned)(size_t)ldst),
                 "v"((unsigned long long)(size_t)gsrc)
               : "memory");
}
__device__ __forceinline__ void wait_async0() {
  asm volatile("s_wait_asynccnt 0x0" ::: "memory");
}

// ---------------------------------------------------------------------------
// f32 -> bf16 bulk convert (weights, context)
// ---------------------------------------------------------------------------
__global__ __launch_bounds__(256) void cvt_bf16_kernel(const float* __restrict__ src,
                                                       u16* __restrict__ dst, size_t n4) {
  size_t i = (size_t)blockIdx.x * 256 + threadIdx.x;
  if (i >= n4) return;
  float4 v = *(const float4*)(src + i * 4);
  uint2 o;
  o.x = f2bf(v.x) | (f2bf(v.y) << 16);
  o.y = f2bf(v.z) | (f2bf(v.w) << 16);
  *(uint2*)(dst + i * 4) = o;
}

// ---------------------------------------------------------------------------
// LayerNorm over C=1024, f32 in -> bf16 out. One 256-thread block per row.
// ---------------------------------------------------------------------------
__global__ __launch_bounds__(256) void ln_kernel(const float* __restrict__ x,
                                                 const float* __restrict__ g,
                                                 const float* __restrict__ bta,
                                                 u16* __restrict__ out) {
  const int t = threadIdx.x;
  const float* xr = x + (size_t)blockIdx.x * C_;
  float4 v = *(const float4*)(xr + t * 4);
  float s = v.x + v.y + v.z + v.w;
  float q = v.x * v.x + v.y * v.y + v.z * v.z + v.w * v.w;
#pragma unroll
  for (int m = 16; m >= 1; m >>= 1) {
    s += __shfl_xor(s, m, 32);
    q += __shfl_xor(q, m, 32);
  }
  __shared__ float ss[8], qq[8];
  if ((t & 31) == 0) { ss[t >> 5] = s; qq[t >> 5] = q; }
  __syncthreads();
  s = 0.f; q = 0.f;
#pragma unroll
  for (int i = 0; i < 8; ++i) { s += ss[i]; q += qq[i]; }
  const float mean = s * (1.f / C_);
  const float var  = q * (1.f / C_) - mean * mean;
  const float rstd = rsqrtf(var + 1e-5f);
  const int c = t * 4;
  const float4 gg = *(const float4*)(g + c);
  const float4 bb = *(const float4*)(bta + c);
  uint2 o;
  o.x = f2bf((v.x - mean) * rstd * gg.x + bb.x) | (f2bf((v.y - mean) * rstd * gg.y + bb.y) << 16);
  o.y = f2bf((v.z - mean) * rstd * gg.z + bb.z) | (f2bf((v.w - mean) * rstd * gg.w + bb.w) << 16);
  *(uint2*)(out + (size_t)blockIdx.x * C_ + c) = o;
}

// ---------------------------------------------------------------------------
// Row softmax over 2048 bf16 elements, in-place. One block per row.
// ---------------------------------------------------------------------------
__global__ __launch_bounds__(256) void softmax2048_kernel(u16* __restrict__ S) {
  const int t = threadIdx.x;
  const size_t base = (size_t)blockIdx.x * 2048;
  uint4 d = *(const uint4*)(S + base + t * 8);
  float v[8];
  v[0] = bf2f(d.x & 0xffffu); v[1] = bf2f(d.x >> 16);
  v[2] = bf2f(d.y & 0xffffu); v[3] = bf2f(d.y >> 16);
  v[4] = bf2f(d.z & 0xffffu); v[5] = bf2f(d.z >> 16);
  v[6] = bf2f(d.w & 0xffffu); v[7] = bf2f(d.w >> 16);
  __shared__ float red[8];
  float mx = v[0];
#pragma unroll
  for (int i = 1; i < 8; ++i) mx = fmaxf(mx, v[i]);
#pragma unroll
  for (int m = 16; m >= 1; m >>= 1) mx = fmaxf(mx, __shfl_xor(mx, m, 32));
  if ((t & 31) == 0) red[t >> 5] = mx;
  __syncthreads();
  mx = red[0];
#pragma unroll
  for (int i = 1; i < 8; ++i) mx = fmaxf(mx, red[i]);
  __syncthreads();
  float s = 0.f;
#pragma unroll
  for (int i = 0; i < 8; ++i) { v[i] = __expf(v[i] - mx); s += v[i]; }
#pragma unroll
  for (int m = 16; m >= 1; m >>= 1) s += __shfl_xor(s, m, 32);
  if ((t & 31) == 0) red[t >> 5] = s;
  __syncthreads();
  s = 0.f;
#pragma unroll
  for (int i = 0; i < 8; ++i) s += red[i];
  const float inv = 1.f / s;
  d.x = f2bf(v[0] * inv) | (f2bf(v[1] * inv) << 16);
  d.y = f2bf(v[2] * inv) | (f2bf(v[3] * inv) << 16);
  d.z = f2bf(v[4] * inv) | (f2bf(v[5] * inv) << 16);
  d.w = f2bf(v[6] * inv) | (f2bf(v[7] * inv) << 16);
  *(uint4*)(S + base + t * 8) = d;
}

// ---------------------------------------------------------------------------
// Generic batched bf16 GEMM:  Y = epilogue( scale * (A @ W^T) + bias [+ res] )
//   A: [batch, M, K] bf16 row-major (lda)
//   W: [batch, Nout, K] bf16 row-major (ldw)       (i.e. B = W^T)
// Tile: 128x128 per 256-thread block (8 waves, wave -> 64x32, 4x2 wmma frags).
// Double-buffered LDS; k-tiles DMAed with global_load_async_to_lds_b128 so the
// next tile's DMA overlaps this tile's WMMAs.
// ---------------------------------------------------------------------------
#define GF_RELU   1
#define GF_OBF16  2
#define GF_OTRANS 4
#define GF_RES    8

__global__ __launch_bounds__(256) void gemm_bf16_kernel(
    const u16* __restrict__ A, unsigned long long sAb, int lda,
    const u16* __restrict__ W, unsigned long long sWb, int ldw,
    const float* __restrict__ bias,
    const float* __restrict__ res, unsigned long long sRb, int ldr,
    void* __restrict__ outp, unsigned long long sOb, int ldo,
    int K, float scale, int flags) {
  __shared__ __align__(16) u16 As[2][128 * 40];
  __shared__ __align__(16) u16 Bs[2][128 * 40];

  const int lane = threadIdx.x & 31;
  const int wv   = threadIdx.x >> 5;
  const int wm   = (wv & 1) * 64;     // wave M offset in tile
  const int wn   = (wv >> 1) * 32;    // wave N offset in tile
  const int cl   = lane & 15;
  const int hi   = lane >> 4;

  const int n0 = blockIdx.x * 128;
  const int m0 = blockIdx.y * 128;
  const u16* Ab = A + (size_t)blockIdx.z * sAb + (size_t)m0 * lda;
  const u16* Wb = W + (size_t)blockIdx.z * sWb + (size_t)n0 * ldw;

  // staging geometry: this thread owns rows (srow, srow+64) at 16B column sko
  const int srow = threadIdx.x >> 2;       // 0..63
  const int sko  = (threadIdx.x & 3) * 8;  // 0,8,16,24
  const u16* ga0 = Ab + (size_t)srow * lda + sko;
  const u16* ga1 = ga0 + (size_t)64 * lda;
  const u16* gb0 = Wb + (size_t)srow * ldw + sko;
  const u16* gb1 = gb0 + (size_t)64 * ldw;
  const int l0 = srow * 40 + sko;
  const int l1 = (srow + 64) * 40 + sko;

  v8f acc[4][2];
#pragma unroll
  for (int fm = 0; fm < 4; ++fm)
#pragma unroll
    for (int fn = 0; fn < 2; ++fn) acc[fm][fn] = (v8f){0.f, 0.f, 0.f, 0.f, 0.f, 0.f, 0.f, 0.f};

  // prologue: DMA first k-tile into buffer 0
  async_b128(ga0, &As[0][l0]);
  async_b128(ga1, &As[0][l1]);
  async_b128(gb0, &Bs[0][l0]);
  async_b128(gb1, &Bs[0][l1]);
  wait_async0();
  __syncthreads();

  int buf = 0;
  for (int kt = 0; kt < K; kt += 32) {
    // kick off the next tile's DMA while we compute on this one
    if (kt + 32 < K) {
      const int nb = buf ^ 1;
      async_b128(ga0 + kt + 32, &As[nb][l0]);
      async_b128(ga1 + kt + 32, &As[nb][l1]);
      async_b128(gb0 + kt + 32, &Bs[nb][l0]);
      async_b128(gb1 + kt + 32, &Bs[nb][l1]);
    }

    const u16* as = As[buf];
    const u16* bs = Bs[buf];
    BF16Frag af[4], bfr[2];
#pragma unroll
    for (int fm = 0; fm < 4; ++fm) {
      const u16* ap = as + (wm + fm * 16 + cl) * 40;
#pragma unroll
      for (int v = 0; v < 8; ++v) {
        const int kb = 2 * v + hi * 16;   // contiguous k order, matches B
        af[fm].u[v] = *(const unsigned*)(ap + kb);
      }
    }
#pragma unroll
    for (int fn = 0; fn < 2; ++fn) {
      const u16* bp = bs + (wn + fn * 16 + cl) * 40;
#pragma unroll
      for (int v = 0; v < 8; ++v)
        bfr[fn].u[v] = *(const unsigned*)(bp + 2 * v + hi * 16);  // contiguous k order, matches A
    }
#pragma unroll
    for (int fm = 0; fm < 4; ++fm)
#pragma unroll
      for (int fn = 0; fn < 2; ++fn)
        acc[fm][fn] = __builtin_amdgcn_wmma_f32_16x16x32_bf16(
            false, af[fm].v, false, bfr[fn].v, (short)0, acc[fm][fn], false, false);

    wait_async0();     // next tile's DMA has landed in LDS
    __syncthreads();   // everyone done reading this tile
    buf ^= 1;
  }

  // epilogue
  const bool relu   = flags & GF_RELU;
  const bool obf    = flags & GF_OBF16;
  const bool otr    = flags & GF_OTRANS;
  const bool hasres = flags & GF_RES;
  float* outf = (float*)outp;
  u16*   outh = (u16*)outp;
  const size_t ob = (size_t)blockIdx.z * sOb;
  const size_t rb = (size_t)blockIdx.z * sRb;

#pragma unroll
  for (int fm = 0; fm < 4; ++fm) {
#pragma unroll
    for (int fn = 0; fn < 2; ++fn) {
      const int col = n0 + wn + fn * 16 + cl;
      const float bi = bias ? bias[col] : 0.f;
#pragma unroll
      for (int r = 0; r < 8; ++r) {
        const int row = m0 + wm + fm * 16 + r + hi * 8;
        float v = acc[fm][fn][r] * scale + bi;
        if (relu) v = fmaxf(v, 0.f);
        if (hasres) v += res[rb + (size_t)row * ldr + col];
        if (otr)      outh[ob + (size_t)col * ldo + row] = (u16)f2bf(v);
        else if (obf) outh[ob + (size_t)row * ldo + col] = (u16)f2bf(v);
        else          outf[ob + (size_t)row * ldo + col] = v;
      }
    }
  }
}

// ---------------------------------------------------------------------------
// Self-attention with sequence length 16 (axis B) per (n, h).
// One wave per block: QK^T via 2 WMMAs, LDS softmax, P·V via 4 WMMAs.
// qkv: [16, N, 3C] bf16; out: [16, N, C] bf16.
// ---------------------------------------------------------------------------
__global__ __launch_bounds__(32) void selfattn_kernel(const u16* __restrict__ qkv,
                                                      u16* __restrict__ out) {
  __shared__ __align__(16) u16 Qs[16 * 72];
  __shared__ __align__(16) u16 Ks[16 * 72];
  __shared__ __align__(16) u16 Vs[16 * 72];
  __shared__ __align__(16) u16 VT[64 * 32];
  __shared__ __align__(16) float Sm[16 * 17];
  __shared__ __align__(16) u16 PA[16 * 32];

  const int lane = threadIdx.x;
  const int n  = blockIdx.x >> 4;
  const int hh = blockIdx.x & 15;
  const int cl = lane & 15;
  const int hi = lane >> 4;

  // async-DMA Q,K,V (16x64 each) into LDS
#pragma unroll
  for (int it = 0; it < 4; ++it) {
    const int chunk = lane + it * 32;          // 0..127
    const int row = chunk >> 3;
    const int off = (chunk & 7) * 8;
    const size_t gbase = ((size_t)row * N_ + n) * (3 * C_) + hh * HD_ + off;
    async_b128(qkv + gbase,          &Qs[row * 72 + off]);
    async_b128(qkv + gbase + C_,     &Ks[row * 72 + off]);
    async_b128(qkv + gbase + 2 * C_, &Vs[row * 72 + off]);
  }
  wait_async0();
  __syncthreads();

  // zero the padded half of VT (k = 16..31), then transpose V into VT[c][j]
  uint4 z; z.x = 0; z.y = 0; z.z = 0; z.w = 0;
#pragma unroll
  for (int it = 0; it < 4; ++it) {
    const int chunk = lane + it * 32;          // 0..127
    const int c = chunk >> 1, half = chunk & 1;
    *(uint4*)&VT[c * 32 + 16 + half * 8] = z;
  }
  for (int idx = lane; idx < 1024; idx += 32) {
    const int j = idx >> 6, c = idx & 63;
    VT[c * 32 + j] = Vs[j * 72 + c];
  }
  __syncthreads();

  // scores = Q @ K^T / sqrt(64)
  v8f sc = (v8f){0.f, 0.f, 0.f, 0.f, 0.f, 0.f, 0.f, 0.f};
#pragma unroll
  for (int t = 0; t < 2; ++t) {
    BF16Frag qa, kb;
#pragma unroll
    for (int v = 0; v < 8; ++v) {
      const int ka = 2 * v + hi * 16 + t * 32;
      qa.u[v] = *(const unsigned*)&Qs[cl * 72 + ka];
      const int kbb = 2 * v + hi * 16 + t * 32;
      kb.u[v] = *(const unsigned*)&Ks[cl * 72 + kbb];
    }
    sc = __builtin_amdgcn_wmma_f32_16x16x32_bf16(false, qa.v, false, kb.v, (short)0, sc,
                                                 false, false);
  }
#pragma unroll
  for (int r = 0; r < 8; ++r) Sm[(r + 8 * hi) * 17 + cl] = sc[r] * 0.125f;
  __syncthreads();

  // softmax over keys j, rows handled by lanes 0..15
  if (lane < 16) {
    float mx = -1e30f;
    for (int j = 0; j < 16; ++j) mx = fmaxf(mx, Sm[lane * 17 + j]);
    float s = 0.f;
    for (int j = 0; j < 16; ++j) s += __expf(Sm[lane * 17 + j] - mx);
    const float inv = 1.f / s;
    for (int j = 0; j < 16; ++j)
      PA[lane * 32 + j] = (u16)f2bf(__expf(Sm[lane * 17 + j] - mx) * inv);
    for (int j = 16; j < 32; ++j) PA[lane * 32 + j] = 0;
  }
  __syncthreads();

  // out = P @ V   (K padded 16 -> 32 with zeros)
#pragma unroll
  for (int ct = 0; ct < 4; ++ct) {
    BF16Frag pa, vb;
#pragma unroll
    for (int v = 0; v < 8; ++v) {
      const int ka = 2 * v + hi * 16;
      pa.u[v] = *(const unsigned*)&PA[cl * 32 + ka];
      vb.u[v] = *(const unsigned*)&VT[(ct * 16 + cl) * 32 + 2 * v + hi * 16];
    }
    v8f o = (v8f){0.f, 0.f, 0.f, 0.f, 0.f, 0.f, 0.f, 0.f};
    o = __builtin_amdgcn_wmma_f32_16x16x32_bf16(false, pa.v, false, vb.v, (short)0, o,
                                                false, false);
#pragma unroll
    for (int r = 0; r < 8; ++r) {
      const size_t gi = ((size_t)(r + 8 * hi) * N_ + n) * C_ + hh * HD_ + ct * 16 + cl;
      out[gi] = (u16)f2bf(o[r]);
    }
  }
}

// ---------------------------------------------------------------------------
// Host side
// ---------------------------------------------------------------------------
static inline void launch_gemm(hipStream_t s, const u16* A, size_t sA, int lda,
                               const u16* W, size_t sW, int ldw, const float* bias,
                               const float* res, size_t sR, int ldr, void* out, size_t sO,
                               int ldo, int M, int Nout, int K, int batch, float scale,
                               int flags) {
  dim3 g(Nout / 128, M / 128, batch);
  gemm_bf16_kernel<<<g, dim3(256), 0, s>>>(A, (unsigned long long)sA, lda, W,
                                           (unsigned long long)sW, ldw, bias, res,
                                           (unsigned long long)sR, ldr, out,
                                           (unsigned long long)sO, ldo, K, scale, flags);
}

extern "C" void kernel_launch(void* const* d_in, const int* in_sizes, int n_in,
                              void* d_out, int out_size, void* d_ws, size_t ws_size,
                              hipStream_t stream) {
  const float* x        = (const float*)d_in[0];
  const float* ctx      = (const float*)d_in[1];
  const float* ln_g     = (const float*)d_in[2];
  const float* ln_b     = (const float*)d_in[3];
  const float* inproj_w = (const float*)d_in[4];
  const float* inproj_b = (const float*)d_in[5];
  const float* out_w    = (const float*)d_in[6];
  const float* out_b    = (const float*)d_in[7];
  const float* q_w      = (const float*)d_in[8];
  const float* q_b      = (const float*)d_in[9];
  const float* kv_w     = (const float*)d_in[10];
  const float* kv_b     = (const float*)d_in[11];
  const float* ff1_w    = (const float*)d_in[12];
  const float* ff1_b    = (const float*)d_in[13];
  const float* ff2_w    = (const float*)d_in[14];
  const float* ff2_b    = (const float*)d_in[15];

  // workspace layout (bytes): hbf 64MB | scr 256MB | sbuf 128MB | weights ~34MB
  char* p = (char*)d_ws;
  u16* hbf  = (u16*)p;            p += (size_t)BN_ * C_ * 2;           // 67,108,864
  u16* scr  = (u16*)p;            p += (size_t)BN_ * 4 * C_ * 2;       // 268,435,456
  u16* sbuf = (u16*)p;            p += (size_t)BN_ * 2 * C_ * 2;       // 134,217,728 (ctx bf16, then S/P)
  u16* wbf  = (u16*)p;
  u16* w_inproj = wbf;
  u16* w_out    = w_inproj + (size_t)3 * C_ * C_;
  u16* w_q      = w_out + (size_t)C_ * C_;
  u16* w_kv     = w_q + (size_t)C_ * C_;
  u16* w_ff1    = w_kv + (size_t)2 * C_ * 2 * C_;
  u16* w_ff2    = w_ff1 + (size_t)4 * C_ * C_;

  float* xbuf = (float*)d_out;  // residual stream lives in d_out (f32, fully overwritten)

  // cross-attention scratch inside scr
  u16* qcb  = scr;                             // [16, N, C] bf16
  u16* kcb  = scr + (size_t)BN_ * C_;          // [16, N, C] bf16
  u16* vctb = scr + (size_t)2 * BN_ * C_;      // [16, C, N] bf16 (V transposed per batch)

  auto cvt = [&](const float* s, u16* d, size_t n) {
    size_t n4 = n / 4;
    cvt_bf16_kernel<<<dim3((unsigned)((n4 + 255) / 256)), dim3(256), 0, stream>>>(s, d, n4);
  };

  // 1) convert weights + context to bf16
  cvt(inproj_w, w_inproj, (size_t)3 * C_ * C_);
  cvt(out_w, w_out, (size_t)C_ * C_);
  cvt(q_w, w_q, (size_t)C_ * C_);
  cvt(kv_w, w_kv, (size_t)2 * C_ * 2 * C_);
  cvt(ff1_w, w_ff1, (size_t)4 * C_ * C_);
  cvt(ff2_w, w_ff2, (size_t)4 * C_ * C_);
  cvt(ctx, sbuf, (size_t)BN_ * 2 * C_);

  // 2) h = LN(x)
  ln_kernel<<<dim3(BN_), dim3(256), 0, stream>>>(x, ln_g, ln_b, hbf);

  // 3) qkv = h @ in_proj_w^T + b  -> scr [BN, 3C] bf16
  launch_gemm(stream, hbf, 0, C_, w_inproj, 0, C_, inproj_b, nullptr, 0, 0, scr, 0, 3 * C_,
              BN_, 3 * C_, C_, 1, 1.f, GF_OBF16);

  // 4) self-attention (seq=16 per (n,h)) -> hbf [16,N,C] bf16
  selfattn_kernel<<<dim3(N_ * H_), dim3(32), 0, stream>>>(scr, hbf);

  // 5) x1 = x + sa @ out_w^T + out_b  -> xbuf f32
  launch_gemm(stream, hbf, 0, C_, w_out, 0, C_, out_b, x, 0, C_, xbuf, 0, C_, BN_, C_, C_, 1,
              1.f, GF_RES);

  // 6) h2 = LN(x1)
  ln_kernel<<<dim3(BN_), dim3(256), 0, stream>>>(xbuf, ln_g, ln_b, hbf);

  // 7) qc = h2 @ q_w^T + q_b -> qcb bf16
  launch_gemm(stream, hbf, 0, C_, w_q, 0, C_, q_b, nullptr, 0, 0, qcb, 0, C_, BN_, C_, C_, 1,
              1.f, GF_OBF16);

  // 8a) kc = ctx @ kv_w[0:C]^T + kv_b[0:C] -> kcb bf16
  launch_gemm(stream, sbuf, 0, 2 * C_, w_kv, 0, 2 * C_, kv_b, nullptr, 0, 0, kcb, 0, C_, BN_,
              C_, 2 * C_, 1, 1.f, GF_OBF16);
  // 8b) vcT = (ctx @ kv_w[C:2C]^T + kv_b[C:2C])^T per batch -> vctb [16, C, N] bf16
  launch_gemm(stream, sbuf, (size_t)N_ * 2 * C_, 2 * C_, w_kv + (size_t)C_ * 2 * C_, 0,
              2 * C_, kv_b + C_, nullptr, 0, 0, vctb, (size_t)C_ * N_, N_, N_, C_, 2 * C_,
              B_, 1.f, GF_OTRANS);

  // 9) S = (qc @ kc^T) * C^-0.5, batched -> sbuf [16, N, N] bf16 (overwrites ctx copy)
  launch_gemm(stream, qcb, (size_t)N_ * C_, C_, kcb, (size_t)N_ * C_, C_, nullptr, nullptr,
              0, 0, sbuf, (size_t)N_ * N_, N_, N_, N_, C_, B_, 0.03125f, GF_OBF16);

  // 10) P = softmax(S) rows, in place
  softmax2048_kernel<<<dim3(BN_), dim3(256), 0, stream>>>(sbuf);

  // 11) x2 = x1 + P @ V, batched (W = vcT) -> xbuf f32
  launch_gemm(stream, sbuf, (size_t)N_ * N_, N_, vctb, (size_t)C_ * N_, N_, nullptr, xbuf,
              (size_t)N_ * C_, C_, xbuf, (size_t)N_ * C_, C_, N_, C_, N_, B_, 1.f, GF_RES);

  // 12) h3 = LN(x2)
  ln_kernel<<<dim3(BN_), dim3(256), 0, stream>>>(xbuf, ln_g, ln_b, hbf);

  // 13) ffh = relu(h3 @ ff1_w^T + ff1_b) -> scr [BN, 4C] bf16
  launch_gemm(stream, hbf, 0, C_, w_ff1, 0, C_, ff1_b, nullptr, 0, 0, scr, 0, 4 * C_, BN_,
              4 * C_, C_, 1, 1.f, GF_OBF16 | GF_RELU);

  // 14) out = x2 + ffh @ ff2_w^T + ff2_b -> d_out f32
  launch_gemm(stream, scr, 0, 4 * C_, w_ff2, 0, 4 * C_, ff2_b, xbuf, 0, C_, d_out, 0, C_,
              BN_, C_, 4 * C_, 1, 1.f, GF_RES);

  (void)in_sizes; (void)n_in; (void)out_size; (void)ws_size;
}
